// PreEmbedGraphEncoderAdapter_69320772158306
// MI455X (gfx1250) — compile-verified, matrix operand order checked by reading, not run on hardware
//
#include <hip/hip_runtime.h>
#include <hip/hip_bf16.h>

// ---------------- problem constants ----------------
#define Bq 8
#define Nq 8192
#define Cq 256
#define Lq 4
#define Eq 131072
#define Mq (Bq * Nq)          // 65536 total nodes

typedef __bf16    v16bf __attribute__((ext_vector_type(16)));
typedef float     v8f   __attribute__((ext_vector_type(8)));
typedef unsigned  u32x4 __attribute__((ext_vector_type(4)));
typedef int       i32x8 __attribute__((ext_vector_type(8)));

// ---------------- elementwise fp32 -> bf16 cast ----------------
__global__ void cast_f32_bf16(const float* __restrict__ in,
                              __hip_bfloat16* __restrict__ out, int n) {
    int i = blockIdx.x * blockDim.x + threadIdx.x;
    if (i < n) out[i] = __float2bfloat16(in[i]);
}

// ---------------- edge aggregation: agg[dst, coloff + c] += src[srcnode, c] ----
__global__ void aggregate_edges(const __hip_bfloat16* __restrict__ srcfeat,
                                const int* __restrict__ edges,   // [B,2,E]
                                float* __restrict__ agg,
                                int Cagg, int coloff) {
    long long tid = (long long)blockIdx.x * blockDim.x + threadIdx.x;
    const long long total = (long long)Bq * Eq * (Cq / 8);
    if (tid >= total) return;
    int chunk = (int)(tid & 31);            // 32 chunks of 8 channels
    long long eg = tid >> 5;                // global edge id
    int b = (int)(eg / Eq);
    int e = (int)(eg - (long long)b * Eq);
    const int* eb = edges + (size_t)b * 2 * Eq;
    int s = eb[e]       + b * Nq;           // gather index
    int d = eb[Eq + e]  + b * Nq;           // scatter index

    const uint4* p = (const uint4*)(srcfeat + (size_t)s * Cq + chunk * 8);
    uint4 w = *p;                           // 8 bf16
    float* ap = agg + (size_t)d * Cagg + coloff + chunk * 8;
    unsigned wd[4] = {w.x, w.y, w.z, w.w};
    #pragma unroll
    for (int j = 0; j < 4; ++j) {
        float lo = __uint_as_float(wd[j] << 16);          // bf16 -> f32 exact
        float hi = __uint_as_float(wd[j] & 0xFFFF0000u);
        atomicAdd(ap + 2 * j,     lo);
        atomicAdd(ap + 2 * j + 1, hi);
    }
}

// ---------------- TDM: async-load one W k-tile [256 rows x 32 k] into LDS ----
// W is [256, K] bf16 row-major; tile start = &W[0, ks]. TDM walks X (32 elems,
// 64B) fastest then Y (256 rows, stride K elems), packing rows contiguously
// into LDS at lds_off (16KB per tile). Tracked with TENSORcnt.
__device__ __forceinline__ void tdm_load_wtile(unsigned lds_off,
                                               const __hip_bfloat16* gtile,
                                               int K) {
    unsigned long long ga = (unsigned long long)(uintptr_t)gtile;
    u32x4 g0;
    g0[0] = 1u;                                        // count=1, user mode
    g0[1] = lds_off;                                   // lds_addr
    g0[2] = (unsigned)(ga & 0xFFFFFFFFull);            // global_addr lo
    g0[3] = ((unsigned)(ga >> 32) & 0x1FFFFFFu) | (2u << 30); // hi | type=2
    i32x8 g1;
    g1[0] = 0x00010000;            // data_size=1 (2 bytes); no multicast
    g1[1] = (K & 0xFFFF) << 16;    // tensor_dim0 = K
    g1[2] = 256 << 16;             // tensor_dim1 = 256 rows
    g1[3] = 32 << 16;              // tile_dim0 = 32 elems
    g1[4] = 256;                   // tile_dim1 = 256 rows
    g1[5] = K;                     // tensor_dim0_stride = K elems
    g1[6] = 0;
    g1[7] = 0;
    asm volatile("tensor_load_to_lds %0, %1" :: "s"(g0), "s"(g1) : "memory");
}

// ---------------- fused dual-GEMM + bias + ReLU (bf16 WMMA, f32 accum) -------
// D = relu( A_x @ W1^T + A_agg @ W2^T + bias )
// K and concat-ness are compile time: fully unrolled k-loop, TDM-staged
// double-buffered weight tiles in LDS, batched ds_loads, WMMAs back-to-back.
template <int K, bool CONCAT>
__global__ void gemm_dual_bias_relu(const __hip_bfloat16* __restrict__ Ax0,
                                    const __hip_bfloat16* __restrict__ Ax1,
                                    const __hip_bfloat16* __restrict__ Agg,
                                    const __hip_bfloat16* __restrict__ W1,
                                    const __hip_bfloat16* __restrict__ W2,
                                    const float* __restrict__ bias,
                                    float* __restrict__ outF,
                                    __hip_bfloat16* __restrict__ outB) {
    // [buffer][matrix][row(c) * 32 + k] : 2 x 2 x 16KB = 64KB
    __shared__ __hip_bfloat16 sW[2][2][256 * 32];

    const int lane  = threadIdx.x & 31;
    const int w     = threadIdx.x >> 5;
    const int rlane = lane & 15;
    const int khalf = lane >> 4;                 // lanes 0-15: K lo, 16-31: hi
    const int rowBase = blockIdx.x * 32 + (w & 1) * 16;
    const int colBase = (w >> 1) * 64;
    const bool issuer = (w == 0);
    constexpr int STEPS = K / 32;

    v8f acc[4];
    #pragma unroll
    for (int t = 0; t < 4; ++t) acc[t] = (v8f){0.f,0.f,0.f,0.f,0.f,0.f,0.f,0.f};

    // prologue: TDM tile 0, register-preload A fragments for step 0
    if (issuer) {
        tdm_load_wtile((unsigned)(uintptr_t)&sW[0][0][0], W1, K);
        tdm_load_wtile((unsigned)(uintptr_t)&sW[0][1][0], W2, K);
    }
    const int r  = rowBase + rlane;
    const int ka = khalf * 16;
    v16bf a1 = *(const v16bf*)(Ax0 + (size_t)r * 256 + ka);
    v16bf a2 = *(const v16bf*)(Agg + (size_t)r * K + ka);

    #pragma unroll
    for (int it = 0; it < STEPS; ++it) {
        // issue TDM for tile it+1 (its buffer was released by the trailing
        // barrier of step it-1), then wait for tile it (in-order completion).
        if (issuer) {
            if (it + 1 < STEPS) {
                const size_t go = (size_t)(it + 1) * 32;
                tdm_load_wtile((unsigned)(uintptr_t)&sW[(it + 1) & 1][0][0], W1 + go, K);
                tdm_load_wtile((unsigned)(uintptr_t)&sW[(it + 1) & 1][1][0], W2 + go, K);
                __builtin_amdgcn_s_wait_tensorcnt(2);
            } else {
                __builtin_amdgcn_s_wait_tensorcnt(0);
            }
        }
        __syncthreads();                               // tile it visible to all

        // software-pipelined A fragments for step it+1 (compile-time select
        // of the concat half: 32-wide steps never straddle the 256 boundary)
        v16bf a1n = a1, a2n = a2;
        if (it + 1 < STEPS) {
            constexpr bool hiHalfNext = CONCAT;        // evaluated per 'it' below
            const int ksn = (it + 1) * 32;
            const __hip_bfloat16* abase = (CONCAT && ksn >= 256) ? Ax1 : Ax0;
            const int kofs = (CONCAT && ksn >= 256) ? (ksn - 256) : ksn;
            (void)hiHalfNext;
            a1n = *(const v16bf*)(abase + (size_t)r * 256 + kofs + ka);
            a2n = *(const v16bf*)(Agg + (size_t)r * K + ksn + ka);
        }

        // batch all 8 B-fragment LDS loads, then run WMMAs back-to-back
        const __hip_bfloat16* w1t = &sW[it & 1][0][0];
        const __hip_bfloat16* w2t = &sW[it & 1][1][0];
        v16bf b1[4], b2[4];
        #pragma unroll
        for (int t = 0; t < 4; ++t) {
            const int c = colBase + t * 16 + rlane;
            b1[t] = *(const v16bf*)(w1t + c * 32 + ka);   // ds_load_b128 x2
            b2[t] = *(const v16bf*)(w2t + c * 32 + ka);
        }
        #pragma unroll
        for (int t = 0; t < 4; ++t)
            acc[t] = __builtin_amdgcn_wmma_f32_16x16x32_bf16(
                         false, a1, false, b1[t], (short)0, acc[t], false, false);
        #pragma unroll
        for (int t = 0; t < 4; ++t)
            acc[t] = __builtin_amdgcn_wmma_f32_16x16x32_bf16(
                         false, a2, false, b2[t], (short)0, acc[t], false, false);

        a1 = a1n; a2 = a2n;
        __syncthreads();               // release buffer (it&1) for next TDM
    }

    // Epilogue: C/D layout -> lane = N column; VGPR i = row i (+8 for hi lanes)
    #pragma unroll
    for (int t = 0; t < 4; ++t) {
        const int c  = colBase + t * 16 + rlane;
        const float bv = bias[c];
        #pragma unroll
        for (int i = 0; i < 8; ++i) {
            const int rr = rowBase + i + khalf * 8;
            float v = acc[t][i] + bv;
            v = v > 0.f ? v : 0.f;
            outB[(size_t)rr * Cq + c] = __float2bfloat16(v);
            if (outF) outF[(size_t)rr * Cq + c] = v;
        }
    }
}

// ---------------- host-side orchestration ----------------
static inline size_t align256(size_t x) { return (x + 255) & ~(size_t)255; }

extern "C" void kernel_launch(void* const* d_in, const int* in_sizes, int n_in,
                              void* d_out, int out_size, void* d_ws, size_t ws_size,
                              hipStream_t stream) {
    const float* x        = (const float*)d_in[0];   // [B,N,C]
    const float* base_Wr  = (const float*)d_in[1];   // [L,C,C]
    const float* base_Wn  = (const float*)d_in[2];   // [L,C,C]
    const float* base_b   = (const float*)d_in[3];   // [L,C]
    const float* adapt_Wr = (const float*)d_in[4];   // [L,C,2C]
    const float* adapt_Wn = (const float*)d_in[5];   // [L,C,2C]
    const float* adapt_b  = (const float*)d_in[6];   // [L,C]
    const int*   edges    = (const int*)d_in[7];     // [B,2,E]
    (void)in_sizes; (void)n_in; (void)ws_size;

    float* outBase  = (float*)d_out;                        // [M,C]
    float* outAdapt = (float*)d_out + (size_t)Mq * Cq;      // [M,C]

    // workspace carve-up
    char* ws = (char*)d_ws;
    size_t off = 0;
    __hip_bfloat16* latb[Lq + 1];
    for (int i = 0; i <= Lq; ++i) { latb[i] = (__hip_bfloat16*)(ws + off); off = align256(off + (size_t)Mq * Cq * 2); }
    __hip_bfloat16* currb[2];
    for (int i = 0; i < 2; ++i)   { currb[i] = (__hip_bfloat16*)(ws + off); off = align256(off + (size_t)Mq * Cq * 2); }
    float*          aggF = (float*)(ws + off);          off = align256(off + (size_t)Mq * 2 * Cq * 4);
    __hip_bfloat16* aggB = (__hip_bfloat16*)(ws + off); off = align256(off + (size_t)Mq * 2 * Cq * 2);
    __hip_bfloat16* bWr  = (__hip_bfloat16*)(ws + off); off = align256(off + (size_t)Lq * Cq * Cq * 2);
    __hip_bfloat16* bWn  = (__hip_bfloat16*)(ws + off); off = align256(off + (size_t)Lq * Cq * Cq * 2);
    __hip_bfloat16* aWr  = (__hip_bfloat16*)(ws + off); off = align256(off + (size_t)Lq * Cq * 2 * Cq * 2);
    __hip_bfloat16* aWn  = (__hip_bfloat16*)(ws + off); off = align256(off + (size_t)Lq * Cq * 2 * Cq * 2);

    const int TB = 256;
    auto cast = [&](const float* src, __hip_bfloat16* dst, int n) {
        cast_f32_bf16<<<(n + TB - 1) / TB, TB, 0, stream>>>(src, dst, n);
    };

    cast(base_Wr,  bWr, Lq * Cq * Cq);
    cast(base_Wn,  bWn, Lq * Cq * Cq);
    cast(adapt_Wr, aWr, Lq * Cq * 2 * Cq);
    cast(adapt_Wn, aWn, Lq * Cq * 2 * Cq);
    cast(x, latb[0], Mq * Cq);

    const long long aggThreads = (long long)Bq * Eq * (Cq / 8);
    const int aggBlocks = (int)((aggThreads + TB - 1) / TB);
    const int gemmBlocks = Mq / 32;

    // ---- base encoder: lat[i+1] = relu(gconv(lat[i])) ----
    for (int i = 0; i < Lq; ++i) {
        hipMemsetAsync(aggF, 0, (size_t)Mq * Cq * 4, stream);
        aggregate_edges<<<aggBlocks, TB, 0, stream>>>(latb[i], edges, aggF, Cq, 0);
        cast(aggF, aggB, Mq * Cq);
        gemm_dual_bias_relu<256, false><<<gemmBlocks, TB, 0, stream>>>(
            latb[i], nullptr, aggB,
            bWr + (size_t)i * Cq * Cq, bWn + (size_t)i * Cq * Cq,
            base_b + (size_t)i * Cq,
            (i == Lq - 1) ? outBase : nullptr, latb[i + 1]);
    }

    // ---- adapter: curr = relu(gconv(concat(lat[i+1], curr))) ----
    __hip_bfloat16* cur = latb[0];            // curr_0 = x
    for (int i = 0; i < Lq; ++i) {
        __hip_bfloat16* nxt = currb[i & 1];
        hipMemsetAsync(aggF, 0, (size_t)Mq * 2 * Cq * 4, stream);
        aggregate_edges<<<aggBlocks, TB, 0, stream>>>(latb[i + 1], edges, aggF, 2 * Cq, 0);
        aggregate_edges<<<aggBlocks, TB, 0, stream>>>(cur,         edges, aggF, 2 * Cq, Cq);
        cast(aggF, aggB, Mq * 2 * Cq);
        gemm_dual_bias_relu<512, true><<<gemmBlocks, TB, 0, stream>>>(
            latb[i + 1], cur, aggB,
            aWr + (size_t)i * Cq * 2 * Cq, aWn + (size_t)i * Cq * 2 * Cq,
            adapt_b + (size_t)i * Cq,
            (i == Lq - 1) ? outAdapt : nullptr, nxt);
        cur = nxt;
    }
    (void)out_size;
}
